// ConvIntrinsicMedium_9062380995229
// MI455X (gfx1250) — compile-verified, matrix operand order checked by reading, not run on hardware
//
#include <hip/hip_runtime.h>
#include <hip/hip_bf16.h>
#include <stdint.h>

// Problem constants (fixed by the reference)
#define NN      50000
#define RR      5
#define AA      8
#define FF      128
#define TT      2
#define OO      128
#define RA      40
#define KDIM    1024          // AA*FF  : GEMM K
#define NCOLT   128           // AA(k_rot) * TT * (OO/16) : 16-wide column tiles
#define KSTEPS  32            // KDIM / 32
#define LDSTR   1032          // padded bf16 row stride (2064B = 516 dwords; 516%64=4 -> conflict-free)

typedef __attribute__((ext_vector_type(16))) __bf16 v16bf;
typedef __attribute__((ext_vector_type(8)))  float  v8f;

union FragAB { uint4 q[2]; v16bf v; };

static __device__ __forceinline__ unsigned short f32_to_bf16(float x) {
    union { float f; unsigned int u; } c; c.f = x;
    unsigned int u = c.u;
    u += 0x7FFFu + ((u >> 16) & 1u);          // round-to-nearest-even
    return (unsigned short)(u >> 16);
}

// ---------------------------------------------------------------------------
// Prep 1: Cred[a][k] = sum_r interp_coeffs[r][a][k]   (8 x 40)
// ---------------------------------------------------------------------------
__global__ void cred_kernel(const float* __restrict__ ic, float* __restrict__ cred) {
    int i = threadIdx.x;
    if (i < AA * RA) {
        int a = i / RA, k = i % RA;
        float s = 0.f;
#pragma unroll
        for (int r = 0; r < RR; ++r) s += ic[(r * AA + a) * RA + k];
        cred[i] = s;
    }
}

// ---------------------------------------------------------------------------
// Prep 2: pack template_weights into bf16, fragment-major B layout.
// Fragment (ct, s): 16 GEMM columns (col = ct*16 + n16), K in [s*32, s*32+32).
// Lane L (n16 = L&15, half = L>>4) holds 16 bf16:
//   e in 0..7  -> K = s*32 + half*8 + e
//   e in 8..15 -> K = s*32 + 16 + half*8 + (e-8)
// (ISA 7.12.2 16-bit B-matrix striping for wave32 WMMA 16x16x32.)
// col = k_rot*256 + t*128 + o ;  K = a*128 + f ;  W elem = TW[(a+k_rot)%8][t][o][f]
// ---------------------------------------------------------------------------
__global__ void pack_w_kernel(const float* __restrict__ tw, unsigned short* __restrict__ wbf) {
    unsigned id = blockIdx.x * blockDim.x + threadIdx.x;
    if (id >= (unsigned)(NCOLT * KSTEPS * 32 * 16)) return;
    unsigned e    = id & 15u;
    unsigned lane = (id >> 4) & 31u;
    unsigned s    = (id >> 9) & 31u;
    unsigned ct   = id >> 14;
    unsigned half = lane >> 4, n16 = lane & 15u;
    unsigned kk   = s * 32u + (e < 8u ? half * 8u + e : 16u + half * 8u + (e - 8u));
    unsigned krot = ct >> 4;
    unsigned t    = (ct >> 3) & 1u;
    unsigned o    = (ct & 7u) * 16u + n16;
    unsigned a    = kk >> 7, f = kk & 127u;
    unsigned asrc = (a + krot) & 7u;
    float val = tw[(((size_t)asrc * TT + t) * OO + o) * FF + f];
    wbf[id] = f32_to_bf16(val);
}

// ---------------------------------------------------------------------------
// Fused kernel: 16 mesh vertices per block; 256 threads (8 wave32).
// Stage 1: gather + barycentric + Cred mix -> bf16 A-panel (16 x 1024) in LDS.
// Stage 2: WMMA GEMM vs packed W (L2-resident), bias + ReLU + t-sum epilogue.
// ---------------------------------------------------------------------------
__global__ __launch_bounds__(256) void fused_kernel(
        const float* __restrict__ mesh,   // (N,128)
        const float* __restrict__ bw,     // (N,40,3)
        const int*   __restrict__ bidx,   // (N,40,3)
        const float* __restrict__ cred,   // (8,40)
        const unsigned short* __restrict__ wbf, // packed B fragments
        const float* __restrict__ bias,   // (2,128)
        float*       __restrict__ out)    // (N,8,128)
{
    __shared__ __align__(16) unsigned short sA[16 * LDSTR]; // 33 KB A-panel (bf16)
    __shared__ float sCred[AA * RA];

    const int tid = threadIdx.x;
    for (int i = tid; i < AA * RA; i += 256) sCred[i] = cred[i];
    __syncthreads();

    // ---------------- Stage 1: per-thread (n_local, 8 features) ----------------
    const int n_local = tid >> 4;         // 0..15
    const int l16     = tid & 15;         // 0..15
    const int fb      = l16 * 8;          // feature base (8 consecutive f)
    const size_t n    = (size_t)blockIdx.x * 16 + n_local;   // N = 3125*16 exactly

    float acc[AA][8];
#pragma unroll
    for (int a = 0; a < AA; ++a)
#pragma unroll
        for (int i = 0; i < 8; ++i) acc[a][i] = 0.f;

    const float* bwn = bw + n * (RA * 3);
    const int*   bin = bidx + n * (RA * 3);

    for (int k = 0; k < RA; ++k) {
        float sig[8];
#pragma unroll
        for (int i = 0; i < 8; ++i) sig[i] = 0.f;
#pragma unroll
        for (int j = 0; j < 3; ++j) {
            const float w = bwn[k * 3 + j];
            const float4* p = (const float4*)(mesh + (size_t)bin[k * 3 + j] * FF + fb);
            float4 x0 = p[0], x1 = p[1];   // global_load_b128 x2
            sig[0] += w * x0.x; sig[1] += w * x0.y; sig[2] += w * x0.z; sig[3] += w * x0.w;
            sig[4] += w * x1.x; sig[5] += w * x1.y; sig[6] += w * x1.z; sig[7] += w * x1.w;
        }
#pragma unroll
        for (int a = 0; a < AA; ++a) {
            const float ca = sCred[a * RA + k];
#pragma unroll
            for (int i = 0; i < 8; ++i) acc[a][i] += ca * sig[i];
        }
    }

    // Pack S[n_local][a*128 + f] as bf16 into the padded LDS A-panel.
#pragma unroll
    for (int a = 0; a < AA; ++a) {
        unsigned d0 = (unsigned)f32_to_bf16(acc[a][0]) | ((unsigned)f32_to_bf16(acc[a][1]) << 16);
        unsigned d1 = (unsigned)f32_to_bf16(acc[a][2]) | ((unsigned)f32_to_bf16(acc[a][3]) << 16);
        unsigned d2 = (unsigned)f32_to_bf16(acc[a][4]) | ((unsigned)f32_to_bf16(acc[a][5]) << 16);
        unsigned d3 = (unsigned)f32_to_bf16(acc[a][6]) | ((unsigned)f32_to_bf16(acc[a][7]) << 16);
        uint4 q = make_uint4(d0, d1, d2, d3);
        *(uint4*)&sA[n_local * LDSTR + a * FF + fb] = q;   // ds_store_b128
    }
    __syncthreads();

    // ---------------- Stage 2: WMMA GEMM ----------------
    const int wave = tid >> 5;            // 0..7  -> rotation k_rot
    const int lane = tid & 31;
    const int half = lane >> 4;
    const int n16  = lane & 15;
    const int krot = wave;
    const uint4* wq = (const uint4*)wbf;  // 16B granules; fragment = 2 consecutive per lane

    for (int ot = 0; ot < 8; ++ot) {      // 16-wide output (o) tiles
        v8f acc0 = {};                    // t = 0
        v8f acc1 = {};                    // t = 1
        const int ct0 = krot * 16 + ot;
        const int ct1 = ct0 + 8;

#pragma unroll 4
        for (int s = 0; s < KSTEPS; ++s) {
            // A fragment from LDS (ISA 16-bit A layout): two ds_load_b128
            const unsigned short* arow = &sA[n16 * LDSTR + s * 32 + half * 8];
            FragAB af;
            af.q[0] = *(const uint4*)(arow);        // K = kb + half*8 .. +7
            af.q[1] = *(const uint4*)(arow + 16);   // K = kb + 16 + half*8 .. +7

            const size_t b0 = ((size_t)(ct0 * KSTEPS + s) * 32 + lane) * 2;
            const size_t b1 = ((size_t)(ct1 * KSTEPS + s) * 32 + lane) * 2;
            __builtin_prefetch(&wq[b0 + 64], 0, 1);  // next k-step B fragment -> global_prefetch_b8

            FragAB bf0, bf1;
            bf0.q[0] = wq[b0];     bf0.q[1] = wq[b0 + 1];
            bf1.q[0] = wq[b1];     bf1.q[1] = wq[b1 + 1];

            acc0 = __builtin_amdgcn_wmma_f32_16x16x32_bf16(
                       false, af.v, false, bf0.v, (short)0, acc0, false, false);
            acc1 = __builtin_amdgcn_wmma_f32_16x16x32_bf16(
                       false, af.v, false, bf1.v, (short)0, acc1, false, false);
        }

        // Epilogue: z + 40*bias, ReLU, sum over t, store (N,8,128)
        const int ob  = ot * 16 + n16;                 // output channel o
        const float b0v = 40.f * bias[ob];             // t = 0
        const float b1v = 40.f * bias[OO + ob];        // t = 1
        const size_t rowbase = (size_t)blockIdx.x * 16;
#pragma unroll
        for (int r = 0; r < 8; ++r) {                  // C/D layout: M = r + half*8
            const int m = r + half * 8;
            float v = fmaxf(acc0[r] + b0v, 0.f) + fmaxf(acc1[r] + b1v, 0.f);
            out[(rowbase + m) * (AA * OO) + krot * OO + ob] = v;
        }
    }
}

// ---------------------------------------------------------------------------
extern "C" void kernel_launch(void* const* d_in, const int* in_sizes, int n_in,
                              void* d_out, int out_size, void* d_ws, size_t ws_size,
                              hipStream_t stream) {
    const float* mesh = (const float*)d_in[0];   // mesh_signal (N,128)
    const float* bw   = (const float*)d_in[1];   // bary_w (N,5,8,3)
    const float* tw   = (const float*)d_in[2];   // template_weights (8,2,128,128)
    const float* bias = (const float*)d_in[3];   // bias (2,128)
    const float* ic   = (const float*)d_in[4];   // interp_coeffs (5,8,40)
    const int*   bidx = (const int*)d_in[5];     // bary_idx (N,5,8,3)

    const size_t wbf_elems = (size_t)NCOLT * KSTEPS * 32 * 16;   // 2,097,152 bf16 = 4 MB
    unsigned short* wbf = (unsigned short*)d_ws;
    float* cred = (float*)((char*)d_ws + wbf_elems * sizeof(unsigned short));

    cred_kernel<<<1, 320, 0, stream>>>(ic, cred);
    pack_w_kernel<<<(unsigned)((wbf_elems + 255) / 256), 256, 0, stream>>>(tw, wbf);
    fused_kernel<<<NN / 16, 256, 0, stream>>>(mesh, bw, bidx, cred, wbf, bias, (float*)d_out);
}
